// GraspGenerator_23450521436171
// MI455X (gfx1250) — compile-verified, matrix operand order checked by reading, not run on hardware
//
#include <hip/hip_runtime.h>
#include <hip/hip_bf16.h>

// ---------------------------------------------------------------------------
// GraspGenerator for MI455X (gfx1250, wave32, WMMA).
// Strategy: fold BN into f16 weights, pre-pack as WMMA-B fragments,
// run every matmul via v_wmma_f32_16x16x32_f16 (M=16 tiles).
// ---------------------------------------------------------------------------

typedef _Float16 v16h __attribute__((ext_vector_type(16)));
typedef _Float16 v8h  __attribute__((ext_vector_type(8)));
typedef float    v8f  __attribute__((ext_vector_type(8)));

#define Bn   2
#define Nn   20000
#define Sn   1024
#define Cn   256
#define NSn  16
#define RADIUSf 0.05f
#define HMINf  (-0.02f)
#define HMAXf  0.04f
#define EPSf   1e-5f

// layer dims (K padded to multiples of 32 for the WMMA K=32 step)
#define K1P 288   // 259 -> 288 (9 k-tiles)
#define O1  512   // 32 n-tiles
#define O2  256   // 16 n-tiles
#define O3  256   // 16 n-tiles
#define OH  96    // 6 n-tiles (head output)

// ---------------------------------------------------------------------------
// K-index map for 16-bit WMMA fragments (CDNA5 ISA 7.12.2, 16-bit A 16x32):
//  lanes 0-15 : V0-3 -> K=0..7,   V4-7 -> K=16..23
//  lanes 16-31: V0-3 -> K=8..15,  V4-7 -> K=24..31
// Same striping convention used for the pre-packed B fragments.
// ---------------------------------------------------------------------------
__device__ __host__ inline int kmap(int lane, int e) {
  int v = e >> 1, h = e & 1;
  int base = (lane & 16) ? 8 : 0;
  return base + ((v < 4) ? (2 * v) : (8 + 2 * v)) + h;
}

// Load an A-fragment (16xK slice, K-tile kt) from row-major [16][ks] f16 LDS.
// Per lane this is two contiguous 16B chunks (K=base..base+7, base+16..+23).
__device__ inline v16h load_a_frag(const _Float16* lds, int ks, int kt, int lane) {
  int m = lane & 15;
  const _Float16* p = lds + m * ks + kt * 32 + ((lane & 16) ? 8 : 0);
  v8h lo = *(const v8h*)p;
  v8h hi = *(const v8h*)(p + 16);
  return __builtin_shufflevector(lo, hi, 0,1,2,3,4,5,6,7,8,9,10,11,12,13,14,15);
}

// ---------------------------------------------------------------------------
// Kernel: fold BN into weights, cast f16, pack into WMMA-B fragments.
// Fragment layout: frag[(kt*NT + nt)*512 + lane*16 + e]  (one v16h per lane).
// ---------------------------------------------------------------------------
__global__ void pack_weights(const float* __restrict__ W, const float* __restrict__ bias,
                             const float* __restrict__ g, const float* __restrict__ be,
                             _Float16* __restrict__ frag, float* __restrict__ bout,
                             int O, int Ci, int KT, int NT) {
  int tid = blockIdx.x * blockDim.x + threadIdx.x;
  if (tid < O) {
    float sc = g ? g[tid] * rsqrtf(1.f + EPSf) : 1.f;
    bout[tid] = bias[tid] * sc + (be ? be[tid] : 0.f);
  }
  int total = KT * NT * 32;
  if (tid >= total) return;
  int lane = tid & 31;
  int tile = tid >> 5;
  int nt = tile % NT, kt = tile / NT;
  int n = nt * 16 + (lane & 15);
  float sc = g ? g[n] * rsqrtf(1.f + EPSf) : 1.f;
  _Float16* dst = frag + (size_t)tile * 512 + lane * 16;
#pragma unroll
  for (int e = 0; e < 16; ++e) {
    int k = kt * 32 + kmap(lane, e);
    float v = (k < Ci) ? W[(size_t)n * Ci + k] * sc : 0.f;
    dst[e] = (_Float16)v;
  }
}

// ---------------------------------------------------------------------------
// Kernel: ball query (cylinder in view frame), first-16 indices in order.
// 8 waves per block, one wave per (b,s). wave32 ballot + prefix popcount.
// ---------------------------------------------------------------------------
__global__ __launch_bounds__(256) void ballquery(const float* __restrict__ xyz,
                                                 const float* __restrict__ new_xyz,
                                                 const float* __restrict__ view_rot,
                                                 int* __restrict__ idx_out) {
  __shared__ int sidx[8][16];
  int wave = threadIdx.x >> 5;
  int lane = threadIdx.x & 31;
  int gs = blockIdx.x * 8 + wave;
  if (gs >= Bn * Sn) return;
  int b = gs / Sn;
  if (lane < 16) sidx[wave][lane] = 0;

  const float* R = view_rot + (size_t)gs * 9;
  float r00 = R[0], r01 = R[1], r02 = R[2];
  float r10 = R[3], r11 = R[4], r12 = R[5];
  float r20 = R[6], r21 = R[7], r22 = R[8];
  const float* Cp = new_xyz + (size_t)gs * 3;
  float cx = Cp[0], cy = Cp[1], cz = Cp[2];
  float d0 = r00 * cx + r01 * cy + r02 * cz;
  float d1 = r10 * cx + r11 * cy + r12 * cz;
  float d2 = r20 * cx + r21 * cy + r22 * cz;

  const float* xb = xyz + (size_t)b * Nn * 3;
  unsigned lmask = (1u << lane) - 1u;
  int cnt = 0;
  for (int p0 = 0; p0 < Nn && cnt < NSn; p0 += 32) {
    int p = p0 + lane;
    bool m = false;
    if (p < Nn) {
      float px = xb[p * 3], py = xb[p * 3 + 1], pz = xb[p * 3 + 2];
      float xr = r00 * px + r01 * py + r02 * pz - d0;
      float yr = r10 * px + r11 * py + r12 * pz - d1;
      float zr = r20 * px + r21 * py + r22 * pz - d2;
      m = (yr * yr + zr * zr < RADIUSf * RADIUSf) && (xr > HMINf) && (xr < HMAXf);
    }
    unsigned bal = __builtin_amdgcn_ballot_w32(m);
    if (m) {
      int slot = cnt + __popc(bal & lmask);
      if (slot < NSn) sidx[wave][slot] = p;
    }
    cnt += __popc(bal);
  }
  if (lane < NSn) {
    int v = sidx[wave][lane];
    int first = sidx[wave][0];   // ==0 when cnt==0 (matches reference pad rule)
    if (lane >= cnt) v = first;
    idx_out[(size_t)gs * NSn + lane] = v;
  }
}

// ---------------------------------------------------------------------------
// Shared WMMA layer: in_lds [16][ks] f16 -> out_lds [16][NT*16] f16 (ReLU).
// 4 waves split the NT n-tiles; KT k-tiles of v_wmma_f32_16x16x32_f16.
// ---------------------------------------------------------------------------
__device__ inline void wmma_layer(const _Float16* __restrict__ in_lds, int ks,
                                  const _Float16* __restrict__ wfrag,
                                  const float* __restrict__ bias,
                                  int KT, int NT,
                                  _Float16* __restrict__ out_lds, int out_ks,
                                  int lane, int wave) {
  int per = NT >> 2;
  for (int q = 0; q < per; ++q) {
    int nt = wave * per + q;
    float bv = bias[nt * 16 + (lane & 15)];
    v8f acc = {bv, bv, bv, bv, bv, bv, bv, bv};
    for (int kt = 0; kt < KT; ++kt) {
      v16h a = load_a_frag(in_lds, ks, kt, lane);
      const _Float16* wp = wfrag + ((size_t)(kt * NT + nt)) * 512 + lane * 16;
      if (kt + 1 < KT)
        __builtin_prefetch(wfrag + ((size_t)((kt + 1) * NT + nt)) * 512 + lane * 16, 0, 0);
      v16h bm = *(const v16h*)wp;
      acc = __builtin_amdgcn_wmma_f32_16x16x32_f16(false, a, false, bm,
                                                   (short)0, acc, false, false);
    }
    int moff = (lane & 16) ? 8 : 0;
    int n = nt * 16 + (lane & 15);
#pragma unroll
    for (int r = 0; r < 8; ++r) {
      float v = acc[r];
      v = v > 0.f ? v : 0.f;                   // ReLU (BN already folded)
      out_lds[(moff + r) * out_ks + n] = (_Float16)v;
    }
  }
}

// ---------------------------------------------------------------------------
// Kernel: per-(b,s) gather + 3-layer MLP + maxpool over the 16 points.
// One 128-thread workgroup (4 waves) per (b,s).
// ---------------------------------------------------------------------------
__global__ __launch_bounds__(128) void group_mlp(const float* __restrict__ xyz,
                                                 const float* __restrict__ new_xyz,
                                                 const float* __restrict__ view_rot,
                                                 const float* __restrict__ features,
                                                 const int* __restrict__ idx,
                                                 const _Float16* __restrict__ w1,
                                                 const _Float16* __restrict__ w2,
                                                 const _Float16* __restrict__ w3,
                                                 const float* __restrict__ b1,
                                                 const float* __restrict__ b2,
                                                 const float* __restrict__ b3,
                                                 _Float16* __restrict__ pooled) {
  __shared__ _Float16 h0[16 * K1P];
  __shared__ _Float16 h1[16 * O1];
  __shared__ _Float16 h2[16 * O2];
  __shared__ int pidx[16];

  int gs = blockIdx.x;
  int b = gs / Sn;
  int t = threadIdx.x, lane = t & 31, wave = t >> 5;

  if (t < 16) pidx[t] = idx[(size_t)gs * NSn + t];
  __syncthreads();

  // gather 16 x 256 features (channels 3..258 of h0)
  const float* fb = features + (size_t)b * Cn * Nn;
  for (int i = t; i < 16 * Cn; i += 128) {
    int j = i & 15, c = i >> 4;
    h0[j * K1P + 3 + c] = (_Float16)fb[(size_t)c * Nn + pidx[j]];
  }
  // rot_xyz (channels 0..2) = ((g_xyz - center)/R) . view_rot[k][j]
  if (t < 16) {
    const float* P = xyz + ((size_t)b * Nn + pidx[t]) * 3;
    const float* Cp = new_xyz + (size_t)gs * 3;
    const float* R = view_rot + (size_t)gs * 9;
    float dx = (P[0] - Cp[0]) * (1.f / RADIUSf);
    float dy = (P[1] - Cp[1]) * (1.f / RADIUSf);
    float dz = (P[2] - Cp[2]) * (1.f / RADIUSf);
    h0[t * K1P + 0] = (_Float16)(dx * R[0] + dy * R[3] + dz * R[6]);
    h0[t * K1P + 1] = (_Float16)(dx * R[1] + dy * R[4] + dz * R[7]);
    h0[t * K1P + 2] = (_Float16)(dx * R[2] + dy * R[5] + dz * R[8]);
  }
  // zero-pad channels 259..287
  for (int i = t; i < 16 * (K1P - 259); i += 128) {
    int j = i / (K1P - 259), q = i % (K1P - 259);
    h0[j * K1P + 259 + q] = (_Float16)0.f;
  }
  __syncthreads();

  wmma_layer(h0, K1P, w1, b1, K1P / 32, O1 / 16, h1, O1, lane, wave);  // 259->512
  __syncthreads();
  wmma_layer(h1, O1, w2, b2, O1 / 32, O2 / 16, h2, O2, lane, wave);    // 512->256
  __syncthreads();

  // layer3 (256->256) fused with ReLU + maxpool over the 16 points
  for (int q = 0; q < 4; ++q) {
    int nt = wave * 4 + q;
    float bv = b3[nt * 16 + (lane & 15)];
    v8f acc = {bv, bv, bv, bv, bv, bv, bv, bv};
    for (int kt = 0; kt < O2 / 32; ++kt) {
      v16h a = load_a_frag(h2, O2, kt, lane);
      v16h bm = *(const v16h*)(w3 + ((size_t)(kt * (O3 / 16) + nt)) * 512 + lane * 16);
      acc = __builtin_amdgcn_wmma_f32_16x16x32_f16(false, a, false, bm,
                                                   (short)0, acc, false, false);
    }
    float pm = 0.f;                      // ReLU floor doubles as pool identity
#pragma unroll
    for (int r = 0; r < 8; ++r) pm = fmaxf(pm, acc[r]);
    pm = fmaxf(pm, __shfl_xor(pm, 16, 32));   // combine M=0..7 with M=8..15
    if (lane < 16) pooled[(size_t)gs * O3 + nt * 16 + lane] = (_Float16)pm;
  }
}

// ---------------------------------------------------------------------------
// Kernel: head GEMMs batched over s: 16 rows per M-tile, c1/c2 (ReLU) + c3.
// ---------------------------------------------------------------------------
__global__ __launch_bounds__(128) void head(const _Float16* __restrict__ pooled,
                                            const _Float16* __restrict__ wc1,
                                            const _Float16* __restrict__ wc2,
                                            const _Float16* __restrict__ wc3,
                                            const float* __restrict__ bc1,
                                            const float* __restrict__ bc2,
                                            const float* __restrict__ bc3,
                                            float* __restrict__ out) {
  __shared__ _Float16 x0[16 * 256];
  __shared__ _Float16 x1[16 * 256];
  __shared__ _Float16 x2[16 * 256];
  int mt = blockIdx.x;                       // 128 tiles of 16 rows
  int t = threadIdx.x, lane = t & 31, wave = t >> 5;

  const int4* s4 = (const int4*)(pooled + (size_t)mt * 16 * 256);
  int4* d4 = (int4*)x0;
  for (int i = t; i < 512; i += 128) d4[i] = s4[i];
  __syncthreads();

  wmma_layer(x0, 256, wc1, bc1, 8, 16, x1, 256, lane, wave);
  __syncthreads();
  wmma_layer(x1, 256, wc2, bc2, 8, 16, x2, 256, lane, wave);
  __syncthreads();

  // c3: 256 -> 96, no ReLU, f32 out
  for (int nt = wave; nt < OH / 16; nt += 4) {
    float bv = bc3[nt * 16 + (lane & 15)];
    v8f acc = {bv, bv, bv, bv, bv, bv, bv, bv};
    for (int kt = 0; kt < 8; ++kt) {
      v16h a = load_a_frag(x2, 256, kt, lane);
      v16h bm = *(const v16h*)(wc3 + ((size_t)(kt * (OH / 16) + nt)) * 512 + lane * 16);
      acc = __builtin_amdgcn_wmma_f32_16x16x32_f16(false, a, false, bm,
                                                   (short)0, acc, false, false);
    }
    int moff = (lane & 16) ? 8 : 0;
    int n = nt * 16 + (lane & 15);
#pragma unroll
    for (int r = 0; r < 8; ++r)
      out[(size_t)(mt * 16 + moff + r) * OH + n] = acc[r];
  }
}

// ---------------------------------------------------------------------------
// Host launcher
// ---------------------------------------------------------------------------
extern "C" void kernel_launch(void* const* d_in, const int* in_sizes, int n_in,
                              void* d_out, int out_size, void* d_ws, size_t ws_size,
                              hipStream_t stream) {
  (void)in_sizes; (void)n_in; (void)out_size; (void)ws_size;
  const float* xyz      = (const float*)d_in[0];
  const float* new_xyz  = (const float*)d_in[1];
  const float* view_rot = (const float*)d_in[2];
  const float* features = (const float*)d_in[3];
  const float* m1w = (const float*)d_in[4],  *m1b = (const float*)d_in[5];
  const float* m1g = (const float*)d_in[6],  *m1be = (const float*)d_in[7];
  const float* m2w = (const float*)d_in[8],  *m2b = (const float*)d_in[9];
  const float* m2g = (const float*)d_in[10], *m2be = (const float*)d_in[11];
  const float* m3w = (const float*)d_in[12], *m3b = (const float*)d_in[13];
  const float* m3g = (const float*)d_in[14], *m3be = (const float*)d_in[15];
  const float* c1w = (const float*)d_in[16], *c1b = (const float*)d_in[17];
  const float* c1g = (const float*)d_in[18], *c1be = (const float*)d_in[19];
  const float* c2w = (const float*)d_in[20], *c2b = (const float*)d_in[21];
  const float* c2g = (const float*)d_in[22], *c2be = (const float*)d_in[23];
  const float* c3w = (const float*)d_in[24], *c3b = (const float*)d_in[25];
  float* out = (float*)d_out;

  // carve workspace (~2.1 MB total)
  char* ws = (char*)d_ws;
  size_t off = 0;
  auto take = [&](size_t bytes) -> void* {
    void* p = ws + off;
    off += (bytes + 255) & ~(size_t)255;
    return p;
  };
  _Float16* w1f = (_Float16*)take((size_t)9 * 32 * 512 * 2);
  _Float16* w2f = (_Float16*)take((size_t)16 * 16 * 512 * 2);
  _Float16* w3f = (_Float16*)take((size_t)8 * 16 * 512 * 2);
  _Float16* c1f = (_Float16*)take((size_t)8 * 16 * 512 * 2);
  _Float16* c2f = (_Float16*)take((size_t)8 * 16 * 512 * 2);
  _Float16* c3f = (_Float16*)take((size_t)8 * 6 * 512 * 2);
  float* b1 = (float*)take(O1 * 4);
  float* b2 = (float*)take(O2 * 4);
  float* b3 = (float*)take(O3 * 4);
  float* bh1 = (float*)take(256 * 4);
  float* bh2 = (float*)take(256 * 4);
  float* bh3 = (float*)take(OH * 4);
  int* idx = (int*)take((size_t)Bn * Sn * NSn * 4);
  _Float16* pooled = (_Float16*)take((size_t)Bn * Sn * O3 * 2);

  auto pack = [&](const float* W, const float* bias, const float* g, const float* be,
                  _Float16* frag, float* bout, int O, int Ci, int KT, int NT) {
    int total = KT * NT * 32;
    if (O > total) total = O;
    hipLaunchKernelGGL(pack_weights, dim3((total + 127) / 128), dim3(128), 0, stream,
                       W, bias, g, be, frag, bout, O, Ci, KT, NT);
  };
  pack(m1w, m1b, m1g, m1be, w1f, b1, O1, Cn + 3, 9, 32);
  pack(m2w, m2b, m2g, m2be, w2f, b2, O2, O1, 16, 16);
  pack(m3w, m3b, m3g, m3be, w3f, b3, O3, O2, 8, 16);
  pack(c1w, c1b, c1g, c1be, c1f, bh1, 256, 256, 8, 16);
  pack(c2w, c2b, c2g, c2be, c2f, bh2, 256, 256, 8, 16);
  pack(c3w, c3b, nullptr, nullptr, c3f, bh3, OH, 256, 8, 6);

  hipLaunchKernelGGL(ballquery, dim3((Bn * Sn) / 8), dim3(256), 0, stream,
                     xyz, new_xyz, view_rot, idx);

  hipLaunchKernelGGL(group_mlp, dim3(Bn * Sn), dim3(128), 0, stream,
                     xyz, new_xyz, view_rot, features, idx,
                     w1f, w2f, w3f, b1, b2, b3, pooled);

  hipLaunchKernelGGL(head, dim3((Bn * Sn) / 16), dim3(128), 0, stream,
                     pooled, c1f, c2f, c3f, bh1, bh2, bh3, out);
}
